// PolarProjectionDepth_2370821948220
// MI455X (gfx1250) — matrix operand order repaired and checked by reading, not compiled, not run on hardware
//
#include <hip/hip_runtime.h>
#include <hip/hip_bf16.h>

typedef __attribute__((ext_vector_type(16))) __bf16 v16bf;
typedef __attribute__((ext_vector_type(8)))  __bf16 v8bf;
typedef __attribute__((ext_vector_type(8)))  float  v8f;
typedef __attribute__((ext_vector_type(4)))  float  v4f;
typedef __attribute__((ext_vector_type(4)))  int    v4i;

#define B_ 16
#define C_ 128
#define H_ 64
#define W_ 256
#define S_ 32
#define D_ 64
#define WT 4         // w-tile per workgroup -> b128 output stores
#define ASTRIDE 72   // bf16/row: 144B stride -> 16B-aligned rows, conflict-free (36*M mod 64 distinct)

#if defined(__HIP_DEVICE_COMPILE__) && __has_builtin(__builtin_amdgcn_global_load_async_to_lds_b128)
#define USE_ASYNC_LDS 1
#else
#define USE_ASYNC_LDS 0
#endif

#if USE_ASYNC_LDS
typedef __attribute__((address_space(1))) v4i as1_v4i;   // global (AS1) int4
typedef __attribute__((address_space(3))) v4i as3_v4i;   // LDS (AS3) int4
#endif

__global__ __launch_bounds__(256)
void PolarProjectionDepth_fused_kernel(const float* __restrict__ image,
                                       const float* __restrict__ pscales,
                                       const float* __restrict__ camf,
                                       float* __restrict__ out)
{
    __shared__ float  psl[WT][H_ * S_];                    // pixel_scales tiles  32 KB
    __shared__ __align__(32) __bf16 Bt[WT][H_ * D_];       // prob, frag order    32 KB
    __shared__ __align__(16) __bf16 At[WT][C_ * ASTRIDE];  // image tiles bf16    72 KB

    const int w0  = blockIdx.x * WT;
    const int b   = blockIdx.y;
    const int tid = threadIdx.x;

    // image[((b*C + c)*H + h)*W + w] = imgBase + e*W (+wi), e = c*H + h
    const size_t imgBase = (size_t)b * C_ * H_ * W_ + (size_t)w0;

    // Early prefetch of the strided A tile (one line per e covers all 4 w's)
    for (int e = tid; e < C_ * H_; e += 256)
        __builtin_prefetch(&image[imgBase + (size_t)e * W_], 0, 0);

    // ---- Stage 1a: pixel_scales[b,:,w0+wi,:] -> LDS, 16B chunks
    const size_t psBase = ((size_t)b * H_ * W_ + (size_t)w0) * S_;
    for (int q = tid; q < WT * (H_ * S_) / 4; q += 256) {
        const int wi = q & 3;
        const int q2 = q >> 2;
        const int h = q2 >> 3, s4 = (q2 & 7) * 4;
        const float* gsrc = &pscales[psBase + (size_t)wi * S_ + (size_t)h * (W_ * S_) + s4];
        float*       ldst = &psl[wi][h * S_ + s4];
#if USE_ASYNC_LDS
        __builtin_amdgcn_global_load_async_to_lds_b128(
            (as1_v4i*)(uintptr_t)gsrc, (as3_v4i*)(uintptr_t)ldst, 0, 0);
#else
        *(v4f*)ldst = *(const v4f*)gsrc;
#endif
    }

    // ---- Stage 1b: image tiles -> bf16 LDS; one 16B load covers all 4 w's
    for (int e8 = tid; e8 < (C_ * H_) / 8; e8 += 256) {
        const int c  = e8 >> 3;          // 8 chunks per 64-wide row
        const int h0 = (e8 & 7) * 8;
        v8bf p0, p1, p2, p3;
        #pragma unroll
        for (int j = 0; j < 8; ++j) {
            const v4f iv = *(const v4f*)&image[imgBase + (size_t)(c * H_ + h0 + j) * W_];
            p0[j] = (__bf16)iv[0];
            p1[j] = (__bf16)iv[1];
            p2[j] = (__bf16)iv[2];
            p3[j] = (__bf16)iv[3];
        }
        *(v8bf*)&At[0][c * ASTRIDE + h0] = p0;
        *(v8bf*)&At[1][c * ASTRIDE + h0] = p1;
        *(v8bf*)&At[2][c * ASTRIDE + h0] = p2;
        *(v8bf*)&At[3][c * ASTRIDE + h0] = p3;
    }

#if USE_ASYNC_LDS
 #if __has_builtin(__builtin_amdgcn_s_wait_asynccnt)
    __builtin_amdgcn_s_wait_asynccnt(0);
 #else
    asm volatile("s_wait_asynccnt 0" ::: "memory");
 #endif
#endif
    __syncthreads();

    // ---- Stage 2: scores + softmax fully in registers; one thread per (wi, d) column,
    //      all 256 threads active. Write bf16 B tile in fragment order:
    //      element ((kt*4 + nt)*32 + (hlf*16 + n))*16 + e,  h = kt*32 + hlf*8 + (e&7) + (e>>3)*16, d = nt*16 + n
    {
        const int wi = tid >> 6;
        const int d  = tid & 63;
        const float fy    = camf[b * 2 + 1];
        const float stepd = 32.0f - 0.5f * (float)d;        // reversed depth steps
        const float g     = __log2f(fy / stepd) * (2.0f / 9.0f) - 1.0f;
        const float pos   = (g + 1.0f) * 0.5f * (float)(S_ - 1);
        const float i0f   = floorf(pos);
        const float w1    = pos - i0f;
        const int   i0    = (int)i0f;
        const int   i1    = i0 + 1;
        const int   idx0  = min(max(i0, 0), S_ - 1);
        const int   idx1  = min(max(i1, 0), S_ - 1);
        const float c0 = (1.0f - w1) * ((i0 >= 0 && i0 < S_) ? 1.0f : 0.0f);
        const float c1 = w1          * ((i1 >= 0 && i1 < S_) ? 1.0f : 0.0f);

        float sc[H_];
        float mx = -3.402823466e38f;
        #pragma unroll
        for (int h = 0; h < H_; ++h) {
            const float v = psl[wi][h * S_ + idx0] * c0 + psl[wi][h * S_ + idx1] * c1;
            sc[h] = v;
            mx = fmaxf(mx, v);
        }
        float sum = 0.0f;
        #pragma unroll
        for (int h = 0; h < H_; ++h) {
            const float e = __expf(sc[h] - mx);
            sc[h] = e;
            sum += e;
        }
        const float inv = 1.0f / sum;
        const int n = d & 15, nt = d >> 4;
        #pragma unroll
        for (int kt = 0; kt < 2; ++kt) {
            #pragma unroll
            for (int hlf = 0; hlf < 2; ++hlf) {
                v16bf pk;
                #pragma unroll
                for (int e = 0; e < 16; ++e)
                    pk[e] = (__bf16)(sc[kt * 32 + hlf * 8 + (e & 7) + ((e >> 3) << 4)] * inv);
                *(v16bf*)&Bt[wi][((((kt * 4 + nt) * 32) + hlf * 16 + n) << 4)] = pk;
            }
        }
    }
    __syncthreads();

    // ---- Stage 3: per-wave 16x64 strip x 4 w-tiles via v_wmma_f32_16x16x32_bf16
    //      16 independent accumulators -> WMMA RAW distance 16, no hazard nops.
    const int wave = tid >> 5;
    const int lane = tid & 31;
    const int hlf  = lane >> 4;
    const int mrow = lane & 15;
    const int m16  = wave * 16;

    v8f acc[WT][4] = {};

    #pragma unroll
    for (int kt = 0; kt < 2; ++kt) {
        #pragma unroll
        for (int wi = 0; wi < WT; ++wi) {
            // A fragment (16-bit A 16x32): two aligned 16B chunks, K = k0+hlf*8+j and +16
            const __bf16* ap = &At[wi][(m16 + mrow) * ASTRIDE + kt * 32 + hlf * 8];
            const v8bf alo = *(const v8bf*)ap;
            const v8bf ahi = *(const v8bf*)(ap + 16);
            const v16bf afrag = __builtin_shufflevector(alo, ahi,
                0, 1, 2, 3, 4, 5, 6, 7, 8, 9, 10, 11, 12, 13, 14, 15);
            #pragma unroll
            for (int nt = 0; nt < 4; ++nt) {
                const v16bf bfrag = *(const v16bf*)&Bt[wi][(((kt * 4 + nt) * 32) + lane) << 4];
                acc[wi][nt] = __builtin_amdgcn_wmma_f32_16x16x32_bf16(
                    false, afrag, false, bfrag, (short)0, acc[wi][nt], false, false);
            }
        }
    }

    // ---- Store: out[b, c, d, w0..w0+3] as 16B stores; VGPR r -> M = r + 8*hlf, N = mrow
    const size_t outBase = (size_t)b * C_ * D_ * W_ + (size_t)w0;
    #pragma unroll
    for (int nt = 0; nt < 4; ++nt) {
        const int dcol = nt * 16 + mrow;
        #pragma unroll
        for (int r = 0; r < 8; ++r) {
            const int crow = m16 + r + 8 * hlf;
            const v4f st = { acc[0][nt][r], acc[1][nt][r], acc[2][nt][r], acc[3][nt][r] };
            *(v4f*)&out[outBase + ((size_t)crow * D_ + dcol) * W_] = st;
        }
    }
}

extern "C" void kernel_launch(void* const* d_in, const int* in_sizes, int n_in,
                              void* d_out, int out_size, void* d_ws, size_t ws_size,
                              hipStream_t stream) {
    (void)in_sizes; (void)n_in; (void)d_ws; (void)ws_size; (void)out_size;
    const float* image   = (const float*)d_in[0];
    const float* pscales = (const float*)d_in[1];
    const float* camf    = (const float*)d_in[2];
    float*       out     = (float*)d_out;

    dim3 grid(W_ / WT, B_);   // one workgroup per (w-quad, b): 1024 WGs
    PolarProjectionDepth_fused_kernel<<<grid, 256, 0, stream>>>(image, pscales, camf, out);
}
